// IntVOS_63969242907238
// MI455X (gfx1250) — matrix operand order repaired and checked by reading, not compiled, not run on hardware
//
#include <hip/hip_runtime.h>

#define HH 120
#define WW 216
#define CC 100
#define CP 128            // channels padded for K=4x32 WMMA steps
#define PSTRIDE 256       // f16 elements per pixel in split array (8 chunks * 32)
#define WIN 31
#define MAXD 15
#define NOBJ 3

typedef __attribute__((ext_vector_type(16))) _Float16 v16h;
typedef __attribute__((ext_vector_type(8)))  float    v8f;

__device__ __forceinline__ v8f wmma_f16f32(v16h a, v16h b, v8f c) {
  // D = A(16x32 f16) * B(32x16 f16) + C(16x16 f32)
  return __builtin_amdgcn_wmma_f32_16x16x32_f16(
      /*neg_a=*/false, a, /*neg_b=*/false, b,
      /*c_mod=*/(short)0, c, /*reuse_a=*/false, /*reuse_b=*/false);
}

// ---- Prologue 1: exact fp32 squared norms of both embeddings ----
__global__ void norms_kernel(const float* __restrict__ q, const float* __restrict__ p,
                             float* __restrict__ qn, float* __restrict__ pn) {
  int idx = blockIdx.x * blockDim.x + threadIdx.x;
  if (idx >= HH * WW) return;
  const float* qr = q + (size_t)idx * CC;
  const float* pr = p + (size_t)idx * CC;
  float sq = 0.f, sp = 0.f;
  for (int c = 0; c < CC; ++c) {
    float a = qr[c]; sq += a * a;
    float b = pr[c]; sp += b * b;
  }
  qn[idx] = sq;
  pn[idx] = sp;
}

// ---- Prologue 2: hi/lo f16 split of prev embedding, padded 100 -> 128 ----
// Layout: ps[pix][chunk=c/16][ 0..15: hi(c), 16..31: lo(c) ], 64B per chunk.
__global__ void split_kernel(const float* __restrict__ p, _Float16* __restrict__ ps) {
  int idx = blockIdx.x * blockDim.x + threadIdx.x;
  if (idx >= HH * WW * CP) return;
  int pix = idx >> 7;
  int c   = idx & (CP - 1);
  float v = (c < CC) ? p[(size_t)pix * CC + c] : 0.f;
  _Float16 h = (_Float16)v;
  size_t base = (size_t)pix * PSTRIDE + (size_t)(c >> 4) * 32 + (c & 15);
  ps[base]      = h;
  ps[base + 16] = (_Float16)(v - (float)h);
}

// ---- Main: one block per (query row r, 16-wide column tile). 4 waves. ----
__global__ __launch_bounds__(128) void corr_min_kernel(
    const float* __restrict__ q, const _Float16* __restrict__ ps,
    const int* __restrict__ lab, const float* __restrict__ qn,
    const float* __restrict__ pn, float* __restrict__ out)
{
  __shared__ float    s_dot[4][16][48];   // per-wave 16x48 dot strip
  __shared__ float    s_pn[WIN][48];      // prev-norm strip
  __shared__ int      s_lab[WIN][48];     // label strip (-1 = out of bounds)
  __shared__ float    s_qn[16];
  __shared__ unsigned s_red[16][NOBJ];    // order-preserving uint min keys

  const int w0   = blockIdx.x * 16;
  const int r    = blockIdx.y;
  const int tid  = threadIdx.x;
  const int lane = tid & 31;
  const int wave = tid >> 5;
  // Scalar (SGPR) copy of the wave id: makes dy / row-base addressing uniform,
  // so the hot loop uses SALU + saddr-form loads instead of v_mul_u64 chains.
  const int waveS = __builtin_amdgcn_readfirstlane(wave);
  const int m    = lane & 15;   // matrix row/col within tile
  const int kg   = lane >> 4;   // K-group select (WMMA lane layout)

  // ---- preload label / prev-norm strips, query norms, init reduction ----
  for (int idx = tid; idx < WIN * 48; idx += 128) {
    int dyy = idx / 48, pcl = idx - dyy * 48;
    int prr = r + dyy - MAXD;
    int pc  = w0 - 16 + pcl;
    bool inb = (prr >= 0) && (prr < HH) && (pc >= 0) && (pc < WW);
    s_pn[dyy][pcl]  = inb ? pn[prr * WW + pc] : 0.f;
    s_lab[dyy][pcl] = inb ? lab[prr * WW + pc] : -1;
  }
  if (tid < 16) {
    int wc = w0 + tid;
    s_qn[tid] = (wc < WW) ? qn[r * WW + wc] : 0.f;
  }
  if (tid < 16 * NOBJ) s_red[tid / NOBJ][tid % NOBJ] = 0xFFFFFFFFu;
  __syncthreads();

  // ---- A tile (query) once per block: f32 load + hi/lo split in regs ----
  // A k order used here (contiguous, not ISA table order): lanes 0-15 row M=lane, elems 0..15 -> K=0..15;
  // lanes 16-31 same rows, K=16..31 (matches contiguous B chunks).
  int wq = w0 + m; if (wq >= WW) wq = WW - 1;           // clamp; results discarded
  const float* qrow = q + (size_t)(r * WW + wq) * CC;
  v16h a_hi[4], a_lo[4];
#pragma unroll
  for (int ks = 0; ks < 4; ++ks) {
#pragma unroll
    for (int j = 0; j < 8; ++j) {
      int c0 = ks * 32 + 16 * kg + j;
      int c1 = ks * 32 + 16 * kg + 8 + j;
      float v0 = (c0 < CC) ? qrow[c0] : 0.f;
      float v1 = (c1 < CC) ? qrow[c1] : 0.f;
      _Float16 h0 = (_Float16)v0, h1 = (_Float16)v1;
      a_hi[ks][j]     = h0;  a_lo[ks][j]     = (_Float16)(v0 - (float)h0);
      a_hi[ks][j + 8] = h1;  a_lo[ks][j + 8] = (_Float16)(v1 - (float)h1);
    }
  }

  // Per-t per-lane 32-bit offsets into the split array (includes kg chunk).
  int coff[3];
#pragma unroll
  for (int t = 0; t < 3; ++t) {
    int pc  = w0 - 16 + 16 * t + m;
    int pcc = pc < 0 ? 0 : (pc >= WW ? WW - 1 : pc);   // clamp; excluded later
    coff[t] = pcc * PSTRIDE + kg * 32;
  }

  // Valid dy range for this wave (scalar, no divergent continue).
  int lo = MAXD - r; if (lo < 0) lo = 0;               // prr >= 0
  int dyF = waveS;
  if (dyF < lo) dyF += ((lo - dyF + 3) >> 2) << 2;
  int dyE = HH - r + MAXD; if (dyE > WIN) dyE = WIN;   // prr < HH

  float minv[NOBJ] = {1e30f, 1e30f, 1e30f};

  // Step schedule: s = 0..11, t = s%3 (interleaved accumulators), ks = s/3.
  // Software pipeline: 3 B-buffers, prefetch distance 2, carried across dy.
  v16h bh[3], bl[3];
  if (dyF < dyE) {
    const _Float16* rb0 = ps + (size_t)(r + dyF - MAXD) * (WW * PSTRIDE);
    const v16h* v0 = (const v16h*)(rb0 + coff[0]);     // s=0: t=0, ks=0
    bh[0] = v0[0]; bl[0] = v0[1];
    const v16h* v1 = (const v16h*)(rb0 + coff[1]);     // s=1: t=1, ks=0
    bh[1] = v1[0]; bl[1] = v1[1];
  }

  for (int dy = dyF; dy < dyE; dy += 4) {
    const int prr = dy + r - MAXD;                     // scalar
    const _Float16* rb = ps + (size_t)prr * (WW * PSTRIDE);
    const int dyn = dy + 4;
    const int prn = (dyn < dyE) ? (dyn + r - MAXD) : prr;   // scalar select
    const _Float16* rbn = ps + (size_t)prn * (WW * PSTRIDE);

    v8f acc[3] = {};
#pragma unroll
    for (int s = 0; s < 12; ++s) {
      const int t = s % 3, ks = s / 3, b = s % 3;
      // Prefetch step s+2 (wraps into next dy's steps 0/1 at s=10,11).
      {
        const int sp = s + 2;
        const v16h* vp = (sp < 12)
            ? (const v16h*)(rb  + coff[sp % 3] + (sp / 3) * 64)
            : (const v16h*)(rbn + coff[(sp - 12) % 3] + ((sp - 12) / 3) * 64);
        bh[sp % 3] = vp[0];
        bl[sp % 3] = vp[1];
      }
      // 2-term split product: hi*hi + lo*hi + hi*lo  (~fp32 accuracy)
      acc[t] = wmma_f16f32(a_hi[ks], bh[b], acc[t]);
      acc[t] = wmma_f16f32(a_lo[ks], bh[b], acc[t]);
      acc[t] = wmma_f16f32(a_hi[ks], bl[b], acc[t]);
    }

    // Dump C/D tiles to this wave's private LDS strip (same-wave RAW ordered).
    // C/D layout: VGPR i: lanes 0-15 -> (M=i, N=lane); lanes 16-31 -> (M=i+8).
#pragma unroll
    for (int t = 0; t < 3; ++t)
#pragma unroll
      for (int i = 0; i < 8; ++i)
        s_dot[waveS][i + 8 * kg][16 * t + m] = acc[t][i];

    // Extraction + masked running min. lane (m, kg): query col w0+m,
    // dx range [16*kg, 16*kg+16) capped at 31. prev col local = m+1+dx.
    float qv  = s_qn[m];
    int   dx1 = kg ? WIN : 16;
    for (int dx = kg * 16; dx < dx1; ++dx) {
      int pcl = m + 1 + dx;                 // 1..46, inside 48-wide strip
      int obj = s_lab[dy][pcl];             // -1 => OOB, excluded
      float d = qv + s_pn[dy][pcl] - 2.0f * s_dot[waveS][m][pcl];
      if (obj == 0)      minv[0] = fminf(minv[0], d);
      else if (obj == 1) minv[1] = fminf(minv[1], d);
      else if (obj == 2) minv[2] = fminf(minv[2], d);
    }
  }

  // ---- cross-lane / cross-wave reduction via order-preserving uint min ----
#pragma unroll
  for (int o = 0; o < NOBJ; ++o) {
    unsigned u = __float_as_uint(minv[o]);
    unsigned k = (u >> 31) ? ~u : (u | 0x80000000u);
    atomicMin(&s_red[m][o], k);
  }
  __syncthreads();
  if (tid < 16 * NOBJ) {
    int mm = tid / NOBJ, o = tid % NOBJ;
    int wc = w0 + mm;
    if (wc < WW) {
      unsigned k = s_red[mm][o];
      unsigned u = (k & 0x80000000u) ? (k & 0x7FFFFFFFu) : ~k;
      float v = __uint_as_float(u);
      out[(r * WW + wc) * NOBJ + o] = fminf(v, 1.0f);  // where(mask,d,1) cap
    }
  }
}

extern "C" void kernel_launch(void* const* d_in, const int* in_sizes, int n_in,
                              void* d_out, int out_size, void* d_ws, size_t ws_size,
                              hipStream_t stream) {
  (void)in_sizes; (void)n_in; (void)out_size; (void)ws_size;
  const float* q   = (const float*)d_in[0];   // query_embedding  [H,W,C] f32
  const float* p   = (const float*)d_in[1];   // prev_frame_embedding
  const int*   lab = (const int*)d_in[2];     // prev_frame_labels [H,W,1] i32
  // d_in[3] = gt_ids = arange(3): identity mapping, handled implicitly.

  // Workspace layout (~13.5 MB total):
  float*    qn = (float*)d_ws;                         // H*W f32
  float*    pn = qn + HH * WW;                         // H*W f32
  _Float16* psplit = (_Float16*)(pn + HH * WW);        // H*W*256 f16 (hi|lo)

  norms_kernel<<<(HH * WW + 255) / 256, 256, 0, stream>>>(q, p, qn, pn);
  split_kernel<<<(HH * WW * CP + 255) / 256, 256, 0, stream>>>(p, psplit);

  dim3 grid((WW + 15) / 16, HH);              // 14 x 120 blocks
  corr_min_kernel<<<grid, 128, 0, stream>>>(q, psplit, lab, qn, pn,
                                            (float*)d_out);
}
